// MultiHeadAttentionLayer_59734405153227
// MI455X (gfx1250) — matrix operand order repaired and checked by reading, not compiled, not run on hardware
//
#include <hip/hip_runtime.h>
#include <hip/hip_bf16.h>

#define HIDDEN   1024
#define HEADS    16
#define HEAD_DIM 64
#define BATCH    4
#define SEQ      2048
#define MTOT     (BATCH * SEQ)   // 8192

typedef __bf16 bf16;
typedef __attribute__((ext_vector_type(16))) __bf16 v16bf;
typedef __attribute__((ext_vector_type(8)))  __bf16 v8bf;
typedef __attribute__((ext_vector_type(8)))  float  v8f;
typedef __attribute__((ext_vector_type(8)))  float  f32x8;

union V16BF { v16bf v; v8bf h[2]; unsigned short u[16]; };
union F8    { v8f v; f32x8 e; float f[8]; };

__device__ __forceinline__ unsigned short f2bf(float f) {
    unsigned int u = __builtin_bit_cast(unsigned int, f);
    unsigned int r = u + 0x7FFFu + ((u >> 16) & 1u);   // round-to-nearest-even
    return (unsigned short)(r >> 16);
}
__device__ __forceinline__ float bf2f(unsigned short b) {
    return __builtin_bit_cast(float, (unsigned int)b << 16);
}

// ---- DPP butterfly reductions over each 16-lane half (no LDS traffic) ------
// quad_perm xor1 = 0xB1, quad_perm xor2 = 0x4E, row_half_mirror (^7) = 0x141,
// row_mirror (^15) = 0x140.  All controls stay inside a 16-lane row, so every
// source lane is valid: old=0 + bound_ctrl=1 lowers to a bare v_mov_b32_dpp.
template <int CTRL>
__device__ __forceinline__ float dpp_max_step(float x) {
    int s = __builtin_bit_cast(int, x);
    int p = __builtin_amdgcn_update_dpp(0, s, CTRL, 0xf, 0xf, true);
    return fmaxf(x, __builtin_bit_cast(float, p));
}
template <int CTRL>
__device__ __forceinline__ float dpp_add_step(float x) {
    int s = __builtin_bit_cast(int, x);
    int p = __builtin_amdgcn_update_dpp(0, s, CTRL, 0xf, 0xf, true);
    return x + __builtin_bit_cast(float, p);
}
__device__ __forceinline__ float red16_max(float x) {
    x = dpp_max_step<0xB1>(x);
    x = dpp_max_step<0x4E>(x);
    x = dpp_max_step<0x141>(x);
    x = dpp_max_step<0x140>(x);
    return x;
}
__device__ __forceinline__ float red16_sum(float x) {
    x = dpp_add_step<0xB1>(x);
    x = dpp_add_step<0x4E>(x);
    x = dpp_add_step<0x141>(x);
    x = dpp_add_step<0x140>(x);
    return x;
}

// ---- async global->LDS copy (CDNA5, ASYNCcnt-tracked) ----------------------
__device__ __forceinline__ void async_copy_b128(const void* gptr, void* lptr) {
    unsigned long long ga = (unsigned long long)(size_t)gptr;
    // flat LDS aperture: addr[31:0] is the LDS byte offset
    unsigned int la = (unsigned int)(size_t)lptr;
    asm volatile("global_load_async_to_lds_b128 %0, %1, off"
                 :: "v"(la), "v"(ga) : "memory");
}

// ---------------------------------------------------------------- weight cast
__global__ void cast_f32_to_bf16(const float* __restrict__ src,
                                 unsigned short* __restrict__ dst, int n) {
    int i = blockIdx.x * blockDim.x + threadIdx.x;
    if (i < n) dst[i] = f2bf(src[i]);
}

// -------------------------------------------------------------------- GEMM
// C[M,N] = A[M,K] @ W[N,K]^T + bias[N]; block = 256 thr (8 waves), block tile
// 256(M) x 64(N); weight tile 64x32 staged to LDS via async copy, dbl-buffered.
#define OUT_F32_ROWMAJOR 0
#define OUT_BF16_BHSD    1   // [B,H,S,Dh] bf16
#define OUT_BF16_BHDS    2   // [B,H,Dh,S] bf16 (transposed, for V)

template <bool A_F32>
__global__ __launch_bounds__(256)
void gemm_xwT(const void* __restrict__ Ap, const bf16* __restrict__ Wb,
              const float* __restrict__ bias, void* __restrict__ Outp,
              int M, int N, int K, int outmode) {
    __shared__ unsigned short Bt[2][64][32];   // [n][k] bf16 bits, 8 KB

    const int tid  = threadIdx.x;
    const int lane = tid & 31;
    const int wave = tid >> 5;
    const int lr   = lane & 15;
    const int hi   = lane >> 4;       // wave half (0/1)
    const int m0   = blockIdx.y * 256 + wave * 32;
    const int n0   = blockIdx.x * 64;

    const float* Af = (const float*)Ap;
    const bf16*  Ab = (const bf16*)Ap;

    // async-stage geometry: one b128 chunk per thread (64 rows x 64B)
    const int srow = tid >> 2;          // 0..63
    const int skof = (tid & 3) * 8;     // 0,8,16,24

    F8 acc[2][4];
#pragma unroll
    for (int t = 0; t < 2; t++)
#pragma unroll
        for (int j = 0; j < 4; j++)
#pragma unroll
            for (int i = 0; i < 8; i++) acc[t][j].f[i] = 0.f;

    const int NK = K / 32;
    // prologue: stage k-step 0 into buffer 0
    async_copy_b128(Wb + (size_t)(n0 + srow) * K + 0 + skof, &Bt[0][srow][skof]);

#pragma unroll 2
    for (int ks = 0; ks < NK; ks++) {
        const int cur = ks & 1;
        const int k0  = ks * 32;
        if (ks + 1 < NK) {
            async_copy_b128(Wb + (size_t)(n0 + srow) * K + (k0 + 32) + skof,
                            &Bt[1 - cur][srow][skof]);
            asm volatile("s_wait_asynccnt 0x1" ::: "memory");
        } else {
            asm volatile("s_wait_asynccnt 0x0" ::: "memory");
        }
        __syncthreads();   // weight tile [cur] resident for all waves

        // ---- A operands (16x32 bf16, lane = row m, 16 contiguous K), direct from global
        V16BF a[2];
#pragma unroll
        for (int t = 0; t < 2; t++) {
            const int row = m0 + t * 16 + lr;
            if (A_F32) {
                F8 r1, r2;
                r1.e = *(const f32x8*)(Af + (size_t)row * K + k0 + hi * 16);
                r2.e = *(const f32x8*)(Af + (size_t)row * K + k0 + hi * 16 + 8);
#pragma unroll
                for (int i = 0; i < 8; i++) {
                    a[t].u[i]     = f2bf(r1.f[i]);
                    a[t].u[8 + i] = f2bf(r2.f[i]);
                }
            } else {
                a[t].h[0] = *(const v8bf*)(Ab + (size_t)row * K + k0 + hi * 16);
                a[t].h[1] = *(const v8bf*)(Ab + (size_t)row * K + k0 + hi * 16 + 8);
            }
        }
        // ---- B operands from LDS (lane = column n, 16 contiguous K)
#pragma unroll
        for (int j = 0; j < 4; j++) {
            V16BF b;
            b.v = *(const v16bf*)&Bt[cur][j * 16 + lr][hi * 16];
#pragma unroll
            for (int t = 0; t < 2; t++)
                acc[t][j].v = __builtin_amdgcn_wmma_f32_16x16x32_bf16(
                    false, a[t].v, false, b.v, (short)0, acc[t][j].v, false, false);
        }
        __syncthreads();   // all reads of [cur] done before it is re-staged
    }

    // ---- epilogue: C layout VGPR i -> row i + hi*8, lane lr -> col
#pragma unroll
    for (int t = 0; t < 2; t++) {
#pragma unroll
        for (int j = 0; j < 4; j++) {
            const int n  = n0 + j * 16 + lr;
            const float bv = bias[n];
#pragma unroll
            for (int i = 0; i < 8; i++) {
                const int m = m0 + t * 16 + hi * 8 + i;
                const float val = acc[t][j].f[i] + bv;
                if (outmode == OUT_F32_ROWMAJOR) {
                    ((float*)Outp)[(size_t)m * N + n] = val;
                } else {
                    const int bb = m >> 11, s = m & (SEQ - 1);
                    const int h  = n >> 6,  d = n & (HEAD_DIM - 1);
                    size_t idx;
                    if (outmode == OUT_BF16_BHSD)
                        idx = (((size_t)(bb * HEADS + h)) * SEQ + s) * HEAD_DIM + d;
                    else
                        idx = (((size_t)(bb * HEADS + h)) * HEAD_DIM + d) * SEQ + s;
                    ((unsigned short*)Outp)[idx] = f2bf(val);
                }
            }
        }
    }
}

// ------------------------------------------------------------- flash attention
// Block = 128 thr (4 waves), all waves share one (b,h); wave owns a 16-query
// block. K/V tiles (32 keys) async-staged to LDS, double-buffered; online
// softmax with DPP reductions; P transposed through per-wave LDS tile.
// 1/sqrt(Dh) = 0.125 is folded into Q at load (exact power-of-two scale).
__global__ __launch_bounds__(128)
void attention_kernel(const bf16* __restrict__ Qh, const bf16* __restrict__ Kh,
                      const bf16* __restrict__ Vt, unsigned short* __restrict__ Xo) {
    __shared__ unsigned short Kt[2][32][64];   // [key][d]  8 KB
    __shared__ unsigned short Vs[2][64][32];   // [d][key]  8 KB
    __shared__ unsigned short ldsP[4][16][32]; // per-wave P tile, 4 KB

    const int tid  = threadIdx.x;
    const int lane = tid & 31;
    const int wave = tid >> 5;
    const int lr   = lane & 15;
    const int hi   = lane >> 4;

    const int qblk = blockIdx.x * 4 + wave;      // 0 .. B*H*(S/16)-1
    const int bh   = qblk >> 7;                  // SEQ/16 == 128 (same for block)
    const int qb   = (qblk & 127) << 4;

    const bf16* Qp = Qh + (size_t)bh * SEQ * HEAD_DIM;
    const bf16* Kp = Kh + (size_t)bh * SEQ * HEAD_DIM;
    const bf16* Vp = Vt + (size_t)bh * HEAD_DIM * SEQ;

    // async-stage geometry: K tile 32x128B = 256 chunks; V tile 64x64B = 256
    const int kc0 = tid,        kr0 = kc0 >> 3, ko0 = (kc0 & 7) * 8;
    const int kc1 = tid + 128,  kr1 = kc1 >> 3, ko1 = (kc1 & 7) * 8;
    const int vc0 = tid,        vr0 = vc0 >> 2, vo0 = (vc0 & 3) * 8;
    const int vc1 = tid + 128,  vr1 = vc1 >> 2, vo1 = (vc1 & 3) * 8;

    // Q in A-layout, pre-scaled by 0.125 (exact: exponent shift only)
    V16BF qa[2];
#pragma unroll
    for (int c = 0; c < 2; c++) {
        const int d = c * 32;
        qa[c].h[0] = *(const v8bf*)(Qp + (size_t)(qb + lr) * HEAD_DIM + d + hi * 16);
        qa[c].h[1] = *(const v8bf*)(Qp + (size_t)(qb + lr) * HEAD_DIM + d + hi * 16 + 8);
#pragma unroll
        for (int i = 0; i < 16; i++)
            qa[c].u[i] = f2bf(bf2f(qa[c].u[i]) * 0.125f);
    }

    float mrow[8], lrow[8];
    F8 o[4];
#pragma unroll
    for (int i = 0; i < 8; i++) { mrow[i] = -1e30f; lrow[i] = 0.f; }
#pragma unroll
    for (int t = 0; t < 4; t++)
#pragma unroll
        for (int i = 0; i < 8; i++) o[t].f[i] = 0.f;

    const int NK = SEQ / 32;      // 64

    // prologue: stage key-block 0 into buffer 0 (4 async ops / thread / stage)
    {
        async_copy_b128(Kp + (size_t)kr0 * HEAD_DIM + ko0, &Kt[0][kr0][ko0]);
        async_copy_b128(Kp + (size_t)kr1 * HEAD_DIM + ko1, &Kt[0][kr1][ko1]);
        async_copy_b128(Vp + (size_t)vr0 * SEQ + vo0,      &Vs[0][vr0][vo0]);
        async_copy_b128(Vp + (size_t)vr1 * SEQ + vo1,      &Vs[0][vr1][vo1]);
    }

#pragma unroll 2
    for (int ks = 0; ks < NK; ks++) {
        const int cur = ks & 1;
        const int kb  = ks * 32;
        if (ks + 1 < NK) {
            const int kn = kb + 32;
            async_copy_b128(Kp + (size_t)(kn + kr0) * HEAD_DIM + ko0, &Kt[1 - cur][kr0][ko0]);
            async_copy_b128(Kp + (size_t)(kn + kr1) * HEAD_DIM + ko1, &Kt[1 - cur][kr1][ko1]);
            async_copy_b128(Vp + (size_t)vr0 * SEQ + kn + vo0,        &Vs[1 - cur][vr0][vo0]);
            async_copy_b128(Vp + (size_t)vr1 * SEQ + kn + vo1,        &Vs[1 - cur][vr1][vo1]);
            asm volatile("s_wait_asynccnt 0x4" ::: "memory");
        } else {
            asm volatile("s_wait_asynccnt 0x0" ::: "memory");
        }
        __syncthreads();   // K/V tile [cur] resident

        // ---- E = (Q/8) @ K^T for 32 keys (two 16x16 tiles), K from LDS
        F8 e0, e1;
#pragma unroll
        for (int i = 0; i < 8; i++) { e0.f[i] = 0.f; e1.f[i] = 0.f; }
#pragma unroll
        for (int c = 0; c < 2; c++) {
            const int d = c * 32;
            V16BF k0t, k1t;
            k0t.v = *(const v16bf*)&Kt[cur][lr][d + hi * 16];
            k1t.v = *(const v16bf*)&Kt[cur][16 + lr][d + hi * 16];
            e0.v = __builtin_amdgcn_wmma_f32_16x16x32_bf16(
                false, qa[c].v, false, k0t.v, (short)0, e0.v, false, false);
            e1.v = __builtin_amdgcn_wmma_f32_16x16x32_bf16(
                false, qa[c].v, false, k1t.v, (short)0, e1.v, false, false);
        }
        // ---- online softmax (row = hi*8 + i; DPP butterfly over 16 lanes)
        float bm[8];
#pragma unroll
        for (int i = 0; i < 8; i++)
            bm[i] = red16_max(fmaxf(e0.f[i], e1.f[i]));
        float alpha[8];
#pragma unroll
        for (int i = 0; i < 8; i++) {
            const float mn = fmaxf(mrow[i], bm[i]);
            alpha[i] = __expf(mrow[i] - mn);
            mrow[i] = mn;
            e0.f[i] = __expf(e0.f[i] - mn);
            e1.f[i] = __expf(e1.f[i] - mn);
        }
#pragma unroll
        for (int i = 0; i < 8; i++) {
            const float rs = red16_sum(e0.f[i] + e1.f[i]);
            lrow[i] = lrow[i] * alpha[i] + rs;
#pragma unroll
            for (int t = 0; t < 4; t++) o[t].f[i] *= alpha[i];
        }
        // ---- P: C-layout f32 -> A-layout bf16 via per-wave LDS tile
#pragma unroll
        for (int i = 0; i < 8; i++) {
            const int r = hi * 8 + i;
            ldsP[wave][r][lr]      = f2bf(e0.f[i]);
            ldsP[wave][r][16 + lr] = f2bf(e1.f[i]);
        }
        V16BF pa;   // in-wave DS ordering guarantees store->load
        pa.h[0] = *(const v8bf*)&ldsP[wave][lr][hi * 16];
        pa.h[1] = *(const v8bf*)&ldsP[wave][lr][hi * 16 + 8];
        // ---- O += P @ V, V from LDS (column n contiguous: [d][key] layout)
#pragma unroll
        for (int t = 0; t < 4; t++) {
            V16BF vb;
            vb.v = *(const v16bf*)&Vs[cur][t * 16 + lr][hi * 16];
            o[t].v = __builtin_amdgcn_wmma_f32_16x16x32_bf16(
                false, pa.v, false, vb.v, (short)0, o[t].v, false, false);
        }
        __syncthreads();   // reads of [cur] done before re-stage
    }

    // ---- normalize and write concat-head output [B,S,HIDDEN] bf16
    const int bb = bh >> 4, h = bh & 15;
#pragma unroll
    for (int i = 0; i < 8; i++) {
        const float inv = 1.f / lrow[i];
        const int q = qb + hi * 8 + i;
#pragma unroll
        for (int t = 0; t < 4; t++) {
            const int col = h * HEAD_DIM + t * 16 + lr;
            Xo[((size_t)(bb * SEQ + q)) * HIDDEN + col] = f2bf(o[t].f[i] * inv);
        }
    }
}

// ---------------------------------------------------------------- launch
extern "C" void kernel_launch(void* const* d_in, const int* in_sizes, int n_in,
                              void* d_out, int out_size, void* d_ws, size_t ws_size,
                              hipStream_t stream) {
    const float* query = (const float*)d_in[0];
    const float* key_  = (const float*)d_in[1];
    const float* value = (const float*)d_in[2];
    const float* Wq = (const float*)d_in[3];
    const float* bq = (const float*)d_in[4];
    const float* Wk = (const float*)d_in[5];
    const float* bk = (const float*)d_in[6];
    const float* Wv = (const float*)d_in[7];
    const float* bv = (const float*)d_in[8];
    const float* Wo = (const float*)d_in[9];
    const float* bo = (const float*)d_in[10];

    char* ws = (char*)d_ws;
    size_t off = 0;
    const size_t WB  = (size_t)HIDDEN * HIDDEN * sizeof(unsigned short);   // 2 MB
    const size_t ACT = (size_t)MTOT * HIDDEN * sizeof(unsigned short);     // 16 MB
    unsigned short* Wq_b = (unsigned short*)(ws + off); off += WB;
    unsigned short* Wk_b = (unsigned short*)(ws + off); off += WB;
    unsigned short* Wv_b = (unsigned short*)(ws + off); off += WB;
    unsigned short* Wo_b = (unsigned short*)(ws + off); off += WB;
    unsigned short* Qh   = (unsigned short*)(ws + off); off += ACT;        // [B,H,S,Dh]
    unsigned short* Kh   = (unsigned short*)(ws + off); off += ACT;        // [B,H,S,Dh]
    unsigned short* Vt   = (unsigned short*)(ws + off); off += ACT;        // [B,H,Dh,S]
    unsigned short* Xa   = (unsigned short*)(ws + off); off += ACT;        // [B,S,HIDDEN]

    const int wN = HIDDEN * HIDDEN;
    cast_f32_to_bf16<<<wN / 256, 256, 0, stream>>>(Wq, Wq_b, wN);
    cast_f32_to_bf16<<<wN / 256, 256, 0, stream>>>(Wk, Wk_b, wN);
    cast_f32_to_bf16<<<wN / 256, 256, 0, stream>>>(Wv, Wv_b, wN);
    cast_f32_to_bf16<<<wN / 256, 256, 0, stream>>>(Wo, Wo_b, wN);

    dim3 ggrid(HIDDEN / 64, MTOT / 256);
    gemm_xwT<true><<<ggrid, 256, 0, stream>>>(query, (const bf16*)Wq_b, bq, Qh,
                                              MTOT, HIDDEN, HIDDEN, OUT_BF16_BHSD);
    gemm_xwT<true><<<ggrid, 256, 0, stream>>>(key_,  (const bf16*)Wk_b, bk, Kh,
                                              MTOT, HIDDEN, HIDDEN, OUT_BF16_BHSD);
    gemm_xwT<true><<<ggrid, 256, 0, stream>>>(value, (const bf16*)Wv_b, bv, Vt,
                                              MTOT, HIDDEN, HIDDEN, OUT_BF16_BHDS);

    attention_kernel<<<BATCH * HEADS * (SEQ / 16) / 4, 128, 0, stream>>>(
        (const bf16*)Qh, (const bf16*)Kh, (const bf16*)Vt, Xa);

    gemm_xwT<false><<<ggrid, 256, 0, stream>>>(Xa, (const bf16*)Wo_b, bo, d_out,
                                               MTOT, HIDDEN, HIDDEN, OUT_F32_ROWMAJOR);
}
